// GCN_24361054503006
// MI455X (gfx1250) — matrix operand order repaired and checked by reading, not compiled, not run on hardware
//
#include <hip/hip_runtime.h>
#include <hip/hip_bf16.h>

#define C_IN  128
#define C_HID 256
#define C_OUT 128
#define LN_EPS 1e-5f

typedef __attribute__((ext_vector_type(16))) __bf16 v16bf;
typedef __attribute__((ext_vector_type(8)))  float  v8f;
typedef __attribute__((ext_vector_type(4)))  unsigned int u32x4;
typedef __attribute__((ext_vector_type(8)))  int          i32x8;
typedef __attribute__((ext_vector_type(4)))  int          i32x4;

union ABFrag { v16bf v; uint4 q[2]; };

__device__ __forceinline__ unsigned short f2bf(float f) {
    unsigned int u = __float_as_uint(f);
    u = (u + 0x7FFFu + ((u >> 16) & 1u)) >> 16;   // round-to-nearest-even
    return (unsigned short)u;
}

// ---------------- conversion kernels ----------------

// W row-major [K x Nout] f32  ->  Wt row-major [Nout x K] bf16 (so B-fragments are contiguous)
__global__ void gcn_cvt_w(const float* __restrict__ W, unsigned short* __restrict__ Wt,
                          int K, int Nout) {
    int idx = blockIdx.x * blockDim.x + threadIdx.x;
    if (idx >= K * Nout) return;
    int k = idx / Nout;
    int n = idx - k * Nout;
    Wt[(size_t)n * K + k] = f2bf(W[idx]);
}

__global__ void gcn_cvt_f2bf(const float* __restrict__ in, unsigned short* __restrict__ out,
                             int n4) {
    int q = blockIdx.x * blockDim.x + threadIdx.x;
    if (q >= n4) return;
    size_t base = (size_t)q * 4;
    float4 v = *(const float4*)(in + base);
    ushort4 o;
    o.x = f2bf(v.x); o.y = f2bf(v.y); o.z = f2bf(v.z); o.w = f2bf(v.w);
    *(ushort4*)(out + base) = o;
}

__global__ void gcn_relu_f2bf(const float* __restrict__ in, unsigned short* __restrict__ out,
                              int n4) {
    int q = blockIdx.x * blockDim.x + threadIdx.x;
    if (q >= n4) return;
    size_t base = (size_t)q * 4;
    float4 v = *(const float4*)(in + base);
    ushort4 o;
    o.x = f2bf(fmaxf(v.x, 0.f)); o.y = f2bf(fmaxf(v.y, 0.f));
    o.z = f2bf(fmaxf(v.z, 0.f)); o.w = f2bf(fmaxf(v.w, 0.f));
    *(ushort4*)(out + base) = o;
}

// ---------------- degree / normalization ----------------

__global__ void gcn_deg_init(float* __restrict__ deg, int n) {
    int i = blockIdx.x * blockDim.x + threadIdx.x;
    if (i < n) deg[i] = 1.0f;            // self-loop
}

__global__ void gcn_deg_count(const int* __restrict__ ei, float* __restrict__ deg, int E) {
    int e = blockIdx.x * blockDim.x + threadIdx.x;
    if (e >= E) return;
    unsafeAtomicAdd(&deg[ei[E + e]], 1.0f);    // in-degree over dst
}

__global__ void gcn_deg_rsqrt(float* __restrict__ deg, int n) {
    int i = blockIdx.x * blockDim.x + threadIdx.x;
    if (i < n) deg[i] = rsqrtf(deg[i]);
}

// ---------------- WMMA GEMM with TDM-staged weights ----------------
// C[M x Ncols] = A[M x K] * Bt[Ncols x K]^T.
// Block = 8 waves, one 64-column group; wave w computes row tile (blockRowGroup*8+w).
// Weight slice (64 rows x K bf16) is DMA'd into LDS once per block by the Tensor Data
// Mover, then B fragments come from LDS (ds_load_b128).
template<int K>
__global__ void gcn_gemm_wmma(const __bf16* __restrict__ A, const __bf16* __restrict__ Bt,
                              float* __restrict__ C, int M, int Ncols) {
    __shared__ unsigned short bsmem[64 * K];

    const int cgs    = Ncols >> 6;
    const int cg     = blockIdx.x % cgs;
    const int mgroup = blockIdx.x / cgs;
    const int wv     = threadIdx.x >> 5;
    const int lane   = threadIdx.x & 31;
    const int mtiles = (M + 15) >> 4;

    // ---- TDM: load 64 x K bf16 weight slice into LDS (wave 0 only) ----
    if (wv == 0) {
        unsigned long long ga = (unsigned long long)(const void*)(Bt + (size_t)(cg * 64) * K);
        unsigned lds = (unsigned)(unsigned long long)(const void*)bsmem;
        u32x4 g0;
        g0[0] = 1u;                                            // count=1 (valid descriptor)
        g0[1] = lds;                                           // lds_addr (bytes)
        g0[2] = (unsigned)ga;                                  // global_addr[31:0]
        g0[3] = ((unsigned)(ga >> 32) & 0x01FFFFFFu) | (2u << 30);  // addr[56:32] | type=2
        i32x8 g1;
        g1[0] = (1 << 16);            // data_size=1 -> 2 bytes; no mask/flags/pad
        g1[1] = (K & 0xFFFF) << 16;   // tensor_dim0[15:0]  (bits 79:48)
        g1[2] = (64 << 16);           // tensor_dim0[31:16]=0 | tensor_dim1[15:0]=64
        g1[3] = (K << 16);            // tensor_dim1[31:16]=0 | tile_dim0=K
        g1[4] = 64;                   // tile_dim1=64, tile_dim2=0
        g1[5] = K;                    // tensor_dim0_stride[31:0] = K elements
        g1[6] = 0;                    // stride0[47:32]=0, stride1[15:0]=0
        g1[7] = 0;                    // stride1[47:16]=0
        i32x4 z4 = {0, 0, 0, 0};      // groups 2/3 disabled (<=2D tensor)
        i32x8 z8 = {0, 0, 0, 0, 0, 0, 0, 0};
        __builtin_amdgcn_tensor_load_to_lds(g0, g1, z4, z4, z8, 0);
        __builtin_amdgcn_s_wait_tensorcnt(0);
    }
    __syncthreads();

    const int mt = mgroup * 8 + wv;
    if (mt >= mtiles) return;

    const int m  = lane & 15;
    const int hi = lane >> 4;

    int rowA = mt * 16 + m;
    if (rowA >= M) rowA = M - 1;                 // clamp (stores guarded below)
    const __bf16* arow = A + (size_t)rowA * K;

    // Preload all A fragments for the whole K loop (one clause of global_load_b128).
    constexpr int KS = K / 32;
    ABFrag afrag[KS];
#pragma unroll
    for (int ks = 0; ks < KS; ++ks) {
        const int k0 = ks * 32;
        afrag[ks].q[0] = *(const uint4*)(arow + k0 + hi * 16);
        afrag[ks].q[1] = *(const uint4*)(arow + k0 + hi * 16 + 8);
    }

    v8f acc[4] = {};
#pragma unroll
    for (int ks = 0; ks < KS; ++ks) {
        const int k0 = ks * 32;
#pragma unroll
        for (int ct = 0; ct < 4; ++ct) {
            // B fragment from LDS: lane = column n, holds K = hi*16 + [0..15] contiguous
            const unsigned short* bp = bsmem + (size_t)(ct * 16 + m) * K + k0 + hi * 16;
            ABFrag b;
            b.q[0] = ((const uint4*)bp)[0];
            b.q[1] = ((const uint4*)bp)[1];
            acc[ct] = __builtin_amdgcn_wmma_f32_16x16x32_bf16(
                false, afrag[ks].v, false, b.v, (short)0, acc[ct], false, false);
        }
    }

    // C layout: lane n = lane&15, VGPR r -> row r + 8*hi
#pragma unroll
    for (int ct = 0; ct < 4; ++ct) {
        const int col = cg * 64 + ct * 16 + m;
#pragma unroll
        for (int r = 0; r < 8; ++r) {
            const int row = mt * 16 + hi * 8 + r;
            if (row < M) C[(size_t)row * Ncols + col] = acc[ct][r];
        }
    }
}

// ---------------- aggregation ----------------

// agg[i,c] = t[i,c]*dinv[i]^2 + bias[c]   (self-loop + bias init)
template<int C>
__global__ void gcn_agg_init(const float* __restrict__ t, const float* __restrict__ dinv,
                             const float* __restrict__ bias, float* __restrict__ agg, int n4) {
    int q = blockIdx.x * blockDim.x + threadIdx.x;
    if (q >= n4) return;
    size_t base = (size_t)q * 4;
    int node = (int)(base / C);
    int c    = (int)(base - (size_t)node * C);
    float s  = dinv[node];
    float s2 = s * s;
    float4 v = *(const float4*)(t + base);
    float4 b = *(const float4*)(bias + c);
    float4 o;
    o.x = v.x * s2 + b.x; o.y = v.y * s2 + b.y;
    o.z = v.z * s2 + b.z; o.w = v.w * s2 + b.w;
    *(float4*)(agg + base) = o;
}

// wave-per-edge scatter: agg[dst] += dinv[src]*dinv[dst] * t[src]
template<int C>
__global__ void gcn_edge_scatter(const int* __restrict__ ei, const float* __restrict__ t,
                                 const float* __restrict__ dinv, float* __restrict__ agg,
                                 int E) {
    const int gw = (blockIdx.x * blockDim.x + threadIdx.x) >> 5;
    if (gw >= E) return;
    const int lane = threadIdx.x & 31;
    const int s = ei[gw];
    const int d = ei[E + gw];
    const float coeff = dinv[s] * dinv[d];
    constexpr int PER = C / 32;                  // 8 (C=256) or 4 (C=128)
    const float* tr = t   + (size_t)s * C + lane * PER;
    float*       ar = agg + (size_t)d * C + lane * PER;
#pragma unroll
    for (int j = 0; j < PER; j += 4) {
        float4 v = *(const float4*)(tr + j);
        unsafeAtomicAdd(ar + j + 0, v.x * coeff);
        unsafeAtomicAdd(ar + j + 1, v.y * coeff);
        unsafeAtomicAdd(ar + j + 2, v.z * coeff);
        unsafeAtomicAdd(ar + j + 3, v.w * coeff);
    }
}

// ---------------- layer norm (wave32 per row, C_OUT = 128) ----------------

__global__ void gcn_layernorm(const float* __restrict__ h, const float* __restrict__ gamma,
                              const float* __restrict__ beta, float* __restrict__ out, int n) {
    const int gw = (blockIdx.x * blockDim.x + threadIdx.x) >> 5;
    if (gw >= n) return;
    const int lane = threadIdx.x & 31;
    const size_t base = (size_t)gw * C_OUT + lane * 4;
    float4 v = *(const float4*)(h + base);

    float s = v.x + v.y + v.z + v.w;
#pragma unroll
    for (int o = 16; o > 0; o >>= 1) s += __shfl_xor(s, o, 32);
    const float mu = s * (1.0f / C_OUT);

    float dx = v.x - mu, dy = v.y - mu, dz = v.z - mu, dw = v.w - mu;
    float ss = dx * dx + dy * dy + dz * dz + dw * dw;
#pragma unroll
    for (int o = 16; o > 0; o >>= 1) ss += __shfl_xor(ss, o, 32);
    const float rs = rsqrtf(ss * (1.0f / C_OUT) + LN_EPS);

    float4 g = *(const float4*)(gamma + lane * 4);
    float4 b = *(const float4*)(beta  + lane * 4);
    float4 o4;
    o4.x = dx * rs * g.x + b.x; o4.y = dy * rs * g.y + b.y;
    o4.z = dz * rs * g.z + b.z; o4.w = dw * rs * g.w + b.w;
    *(float4*)(out + base) = o4;
}

// ---------------- launch ----------------

extern "C" void kernel_launch(void* const* d_in, const int* in_sizes, int n_in,
                              void* d_out, int out_size, void* d_ws, size_t ws_size,
                              hipStream_t stream) {
    const float* x     = (const float*)d_in[0];
    const int*   ei    = (const int*)  d_in[1];
    const float* W1    = (const float*)d_in[2];
    const float* b1    = (const float*)d_in[3];
    const float* W2    = (const float*)d_in[4];
    const float* b2    = (const float*)d_in[5];
    const float* gamma = (const float*)d_in[6];
    const float* beta  = (const float*)d_in[7];
    float* out = (float*)d_out;

    const int N = in_sizes[0] / C_IN;
    const int E = in_sizes[1] / 2;

    // workspace carve-out
    char* w = (char*)d_ws;
    size_t off = 0;
    auto take = [&](size_t bytes) { size_t r = off; off += (bytes + 255) & ~(size_t)255; return r; };
    float*          dinv = (float*)         (w + take((size_t)N * 4));
    unsigned short* w1t  = (unsigned short*)(w + take((size_t)C_IN * C_HID * 2));
    unsigned short* w2t  = (unsigned short*)(w + take((size_t)C_HID * C_OUT * 2));
    unsigned short* xbf  = (unsigned short*)(w + take((size_t)N * C_IN * 2));
    float*          t1   = (float*)         (w + take((size_t)N * C_HID * 4));   // reused as t2
    float*          agg1 = (float*)         (w + take((size_t)N * C_HID * 4));   // reused as agg2
    unsigned short* h1bf = (unsigned short*)(w + take((size_t)N * C_HID * 2));
    float* t2 = t1;
    float* agg2 = agg1;

    const int B = 256;
    auto blk = [](long long t, int b) { return (unsigned)((t + b - 1) / b); };

    // weight / feature conversion
    gcn_cvt_w<<<blk((long long)C_IN * C_HID, B), B, 0, stream>>>(W1, w1t, C_IN, C_HID);
    gcn_cvt_w<<<blk((long long)C_HID * C_OUT, B), B, 0, stream>>>(W2, w2t, C_HID, C_OUT);
    gcn_cvt_f2bf<<<blk((long long)N * (C_IN / 4), B), B, 0, stream>>>(x, xbf, N * (C_IN / 4));

    // symmetric normalization coefficients
    gcn_deg_init<<<blk(N, B), B, 0, stream>>>(dinv, N);
    gcn_deg_count<<<blk(E, B), B, 0, stream>>>(ei, dinv, E);
    gcn_deg_rsqrt<<<blk(N, B), B, 0, stream>>>(dinv, N);

    const int mtiles  = (N + 15) / 16;
    const int mgroups = (mtiles + 7) / 8;

    // layer 1: t1 = x @ W1 (WMMA + TDM-staged weights)
    gcn_gemm_wmma<C_IN><<<mgroups * (C_HID / 64), B, 0, stream>>>(
        (const __bf16*)xbf, (const __bf16*)w1t, t1, N, C_HID);
    gcn_agg_init<C_HID><<<blk((long long)N * (C_HID / 4), B), B, 0, stream>>>(
        t1, dinv, b1, agg1, N * (C_HID / 4));
    gcn_edge_scatter<C_HID><<<blk((long long)E * 32, B), B, 0, stream>>>(ei, t1, dinv, agg1, E);
    gcn_relu_f2bf<<<blk((long long)N * (C_HID / 4), B), B, 0, stream>>>(
        agg1, h1bf, N * (C_HID / 4));

    // layer 2: t2 = relu(h1) @ W2
    gcn_gemm_wmma<C_HID><<<mgroups * (C_OUT / 64), B, 0, stream>>>(
        (const __bf16*)h1bf, (const __bf16*)w2t, t2, N, C_OUT);
    gcn_agg_init<C_OUT><<<blk((long long)N * (C_OUT / 4), B), B, 0, stream>>>(
        t2, dinv, b2, agg2, N * (C_OUT / 4));
    gcn_edge_scatter<C_OUT><<<blk((long long)E * 32, B), B, 0, stream>>>(ei, t2, dinv, agg2, E);

    // layer norm -> output
    gcn_layernorm<<<blk((long long)N * 32, B), B, 0, stream>>>(agg2, gamma, beta, out, N);
}
